// SpatialSelfAttentionBlock_17789754540914
// MI455X (gfx1250) — matrix operand order repaired and checked by reading, not compiled, not run on hardware
//
#include <hip/hip_runtime.h>
#include <hip/hip_bf16.h>

// ---------------------------------------------------------------------------
// SpatialSelfAttentionBlock for MI455X (gfx1250, wave32, WMMA f16->f32)
//   B=8, C=512, H=W=32 (L=1024), 8 heads x 64, GN groups=32
// Pipeline:
//   1) gn_kernel   : GroupNorm, write xn as f16 in [b][l][c] layout
//   2) qkv_kernel  : 3x (512x512x1024) GEMM via v_wmma_f32_16x16x32_f16
//                    Q,K -> [head][l][d] (scale folded in), V -> [head][d][l]
//   3) qk_kernel   : per-head 1024x1024x64 GEMM -> attn_map (f32, to d_out)
//   4) rowstat     : per-row max and 1/sum(exp) (512KB)
//   5) av_kernel   : softmax-on-the-fly PV GEMM -> a f16 [b][l][inner]
//   6) out_kernel  : 512x512x1024 GEMM + bias + residual -> out (f32, d_out)
// All long-K GEMM loops are software-pipelined (register double buffering) so
// fragment loads for iteration k+1 are in flight while iteration k's WMMAs run.
// ---------------------------------------------------------------------------

typedef __attribute__((ext_vector_type(16))) _Float16 v16h;
typedef __attribute__((ext_vector_type(8)))  _Float16 v8h;
typedef __attribute__((ext_vector_type(8)))  float    v8f;
typedef __attribute__((ext_vector_type(4)))  float    v4f;

#define DEV __device__ __forceinline__

static constexpr int BB    = 8;
static constexpr int CC    = 512;
static constexpr int LL    = 1024;   // 32*32
static constexpr int NHEAD = 8;
static constexpr int HD    = 64;
static constexpr int INNER = 512;
static constexpr float QK_SCALE = 0.35355339059327373f;  // 64^-0.25

DEV v8f wmma16(v16h a, v16h b, v8f c) {
  // v_wmma_f32_16x16x32_f16 : D = A(16x32) * B(32x16) + C
  return __builtin_amdgcn_wmma_f32_16x16x32_f16(false, a, false, b,
                                                (short)0, c, false, false);
}

// A fragment (16x32, f16). Lane m = lane&15 holds row m; with koff=(lane>>4)*8
// the lane's halves are k = {koff..koff+7, koff+16..koff+23}. p = rowbase+k0+koff.
DEV v16h load_a16(const _Float16* p) {
  union { v16h v; v8h h[2]; } u;
  u.h[0] = *(const v8h*)(p);
  u.h[1] = *(const v8h*)(p + 16);
  return u.v;
}
DEV v16h load_a_cvt(const float* p) {   // same pattern, f32 source -> f16
  union { v16h v; _Float16 e[16]; } u;
#pragma unroll
  for (int i = 0; i < 8; ++i) {
    u.e[i]     = (_Float16)p[i];
    u.e[8 + i] = (_Float16)p[16 + i];
  }
  return u.v;
}
// B fragment (32x16, f16). Lane n = lane&15 holds column n; with kb=(lane>>4)*8
// the lane's halves are k = {kb..kb+7, kb+16..kb+23}. p = colbase + k0 + kb (col-contig layout).
DEV v16h load_b16(const _Float16* p) {
  union { v16h v; v8h h[2]; } u;
  u.h[0] = *(const v8h*)(p);
  u.h[1] = *(const v8h*)(p + 16);
  return u.v;
}

// ---------------------------------------------------------------------------
// 1) GroupNorm: x (b,c,h,w) f32 -> xnT f16 [b][l][c]
// ---------------------------------------------------------------------------
__global__ void gn_kernel(const float* __restrict__ x,
                          const float* __restrict__ gamma,
                          const float* __restrict__ beta,
                          _Float16* __restrict__ xnT) {
  const int b   = blockIdx.x >> 5;   // /32 groups
  const int g   = blockIdx.x & 31;
  const int tid = threadIdx.x;       // 256
  const float* xg = x + ((size_t)b * CC + g * 16) * LL;  // 16 ch x 1024

  float s = 0.f, s2 = 0.f;
#pragma unroll
  for (int i = 0; i < 64; ++i) {
    float v = xg[tid + (i << 8)];
    s += v; s2 += v * v;
  }
  __shared__ float red[256], red2[256];
  red[tid] = s; red2[tid] = s2;
  __syncthreads();
  for (int off = 128; off > 0; off >>= 1) {
    if (tid < off) { red[tid] += red[tid + off]; red2[tid] += red2[tid + off]; }
    __syncthreads();
  }
  const float mean = red[0] * (1.f / 16384.f);
  const float var  = red2[0] * (1.f / 16384.f) - mean * mean;
  const float rstd = rsqrtf(var + 1e-5f);

  const int cl = tid & 15;
  const int c  = g * 16 + cl;
  const float gm = gamma[c] * rstd;
  const float bt = beta[c] - mean * gm;
#pragma unroll
  for (int i = 0; i < 64; ++i) {
    int l = (tid >> 4) + (i << 4);
    float v = xg[cl * LL + l];
    xnT[(((size_t)b << 10) + l) * CC + c] = (_Float16)(v * gm + bt);
  }
}

// ---------------------------------------------------------------------------
// 2) QKV projections. Per wave: 16(M=o) x 64(N=l) tile, K=c (512, 16 iters),
//    software pipelined.
// ---------------------------------------------------------------------------
__global__ void qkv_kernel(const float* __restrict__ Wq, const float* __restrict__ bq,
                           const float* __restrict__ Wk, const float* __restrict__ bk,
                           const float* __restrict__ Wv, const float* __restrict__ bvp,
                           const _Float16* __restrict__ xnT,
                           _Float16* __restrict__ qo, _Float16* __restrict__ ko,
                           _Float16* __restrict__ vo) {
  const int tid  = threadIdx.x;
  const int lane = tid & 31;
  int wid = (blockIdx.x << 3) + (tid >> 5);     // 0..12287
  const int ngrp  = wid & 15;  wid >>= 4;
  const int mtile = wid & 31;  wid >>= 5;
  const int proj  = wid % 3;
  const int b     = wid / 3;

  const float* W;  const float* bias;
  if      (proj == 0) { W = Wq; bias = bq;  }
  else if (proj == 1) { W = Wk; bias = bk;  }
  else                { W = Wv; bias = bvp; }

  const int m     = lane & 15;
  const int hi    = lane >> 4;
  const int koffA = hi << 3;
  const int kbB   = hi << 3;
  const int n     = lane & 15;
  const int l0    = ngrp << 6;

  const _Float16* xb   = xnT + ((size_t)b << 10) * CC;
  const float*    wrow = W + (size_t)(mtile * 16 + m) * CC + koffA;
  const _Float16* bc0  = xb + (size_t)(l0      + n) * CC + kbB;
  const _Float16* bc1  = xb + (size_t)(l0 + 16 + n) * CC + kbB;
  const _Float16* bc2  = xb + (size_t)(l0 + 32 + n) * CC + kbB;
  const _Float16* bc3  = xb + (size_t)(l0 + 48 + n) * CC + kbB;

  v8f acc0 = {}, acc1 = {}, acc2 = {}, acc3 = {};
  v16h A  = load_a_cvt(wrow);
  v16h B0 = load_b16(bc0), B1 = load_b16(bc1);
  v16h B2 = load_b16(bc2), B3 = load_b16(bc3);
  for (int kk = 0; kk < CC; kk += 32) {
    const int kn = (kk + 32 < CC) ? kk + 32 : kk;   // last iter: harmless reload
    v16h An  = load_a_cvt(wrow + kn);
    v16h B0n = load_b16(bc0 + kn);
    v16h B1n = load_b16(bc1 + kn);
    v16h B2n = load_b16(bc2 + kn);
    v16h B3n = load_b16(bc3 + kn);
    acc0 = wmma16(A, B0, acc0);
    acc1 = wmma16(A, B1, acc1);
    acc2 = wmma16(A, B2, acc2);
    acc3 = wmma16(A, B3, acc3);
    A = An; B0 = B0n; B1 = B1n; B2 = B2n; B3 = B3n;
  }

  const int mbase = mtile * 16 + (hi << 3);        // o of acc element r=0
  union { v8f v; float f[8]; } b8;
  b8.v = *(const v8f*)(bias + mbase);
  const int  head  = mbase >> 6;
  const int  dbase = mbase & 63;
  const size_t bh  = (size_t)b * NHEAD + head;

  union { v8f v; float f[8]; } ua[4];
  ua[0].v = acc0; ua[1].v = acc1; ua[2].v = acc2; ua[3].v = acc3;

  if (proj < 2) {
    _Float16* dst = (proj == 0) ? qo : ko;
#pragma unroll
    for (int t = 0; t < 4; ++t) {
      const int l = l0 + t * 16 + n;
      union { v8h v; _Float16 e[8]; } uh;
#pragma unroll
      for (int r = 0; r < 8; ++r)
        uh.e[r] = (_Float16)((ua[t].f[r] + b8.f[r]) * QK_SCALE);
      *(v8h*)(dst + ((bh << 10) + l) * HD + dbase) = uh.v;
    }
  } else {
#pragma unroll
    for (int t = 0; t < 4; ++t) {
      const int l = l0 + t * 16 + n;
#pragma unroll
      for (int r = 0; r < 8; ++r)
        vo[((bh << 6) + dbase + r) * LL + l] = (_Float16)(ua[t].f[r] + b8.f[r]);
    }
  }
}

// ---------------------------------------------------------------------------
// 3) attn_map = Qs^T Ks per head. Per wave: 16(t) x 64(s), K=d=64 (2 iters,
//    fully unrolled by the compiler).
// ---------------------------------------------------------------------------
__global__ void qk_kernel(const _Float16* __restrict__ q,
                          const _Float16* __restrict__ k,
                          float* __restrict__ attn) {
  const int tid  = threadIdx.x;
  const int lane = tid & 31;
  int wid = (blockIdx.x << 3) + (tid >> 5);     // 0..65535
  const int ngrp  = wid & 15;  wid >>= 4;
  const int mtile = wid & 63;  wid >>= 6;
  const int bh    = wid;                        // 0..63

  const int m = lane & 15, hi = lane >> 4, n = lane & 15;
  const int koffA = hi << 3, kbB = hi << 3;
  const _Float16* qh = q + ((size_t)bh << 10) * HD;
  const _Float16* kh = k + ((size_t)bh << 10) * HD;
  const int t0 = mtile << 4, s0 = ngrp << 6;

  v8f acc0 = {}, acc1 = {}, acc2 = {}, acc3 = {};
#pragma unroll
  for (int kk = 0; kk < HD; kk += 32) {
    v16h A = load_a16(qh + (size_t)(t0 + m) * HD + kk + koffA);
    const _Float16* bp = kh + kk + kbB;
    acc0 = wmma16(A, load_b16(bp + (size_t)(s0      + n) * HD), acc0);
    acc1 = wmma16(A, load_b16(bp + (size_t)(s0 + 16 + n) * HD), acc1);
    acc2 = wmma16(A, load_b16(bp + (size_t)(s0 + 32 + n) * HD), acc2);
    acc3 = wmma16(A, load_b16(bp + (size_t)(s0 + 48 + n) * HD), acc3);
  }

  float* arow = attn + ((size_t)bh << 20);      // 1024*1024 per head
  const int trow = t0 + (hi << 3);
  union { v8f v; float f[8]; } ua[4];
  ua[0].v = acc0; ua[1].v = acc1; ua[2].v = acc2; ua[3].v = acc3;
#pragma unroll
  for (int t = 0; t < 4; ++t)
#pragma unroll
    for (int r = 0; r < 8; ++r)
      arow[(size_t)(trow + r) * LL + s0 + t * 16 + n] = ua[t].f[r];
}

// ---------------------------------------------------------------------------
// 4) Row statistics for softmax: max and 1/sum(exp). One block per row.
// ---------------------------------------------------------------------------
__global__ void rowstat_kernel(const float* __restrict__ attn,
                               float* __restrict__ rmax,
                               float* __restrict__ rrcp) {
  const size_t row = blockIdx.x;                // 65536 rows
  const int tid = threadIdx.x;                  // 256
  const float* p = attn + row * LL;
  v4f x = *(const v4f*)(p + tid * 4);
  float mx = fmaxf(fmaxf(x[0], x[1]), fmaxf(x[2], x[3]));

  __shared__ float red[256];
  red[tid] = mx; __syncthreads();
  for (int off = 128; off > 0; off >>= 1) {
    if (tid < off) red[tid] = fmaxf(red[tid], red[tid + off]);
    __syncthreads();
  }
  mx = red[0];
  __syncthreads();
  float s = __expf(x[0] - mx) + __expf(x[1] - mx) +
            __expf(x[2] - mx) + __expf(x[3] - mx);
  red[tid] = s; __syncthreads();
  for (int off = 128; off > 0; off >>= 1) {
    if (tid < off) red[tid] += red[tid + off];
    __syncthreads();
  }
  if (tid == 0) { rmax[row] = mx; rrcp[row] = 1.f / red[0]; }
}

// ---------------------------------------------------------------------------
// 5) a = softmax(attn) @ V^T, softmax applied on the fly while building A.
//    Per wave: 16(t) x 64(c), K=s (1024, 32 iters), software pipelined;
//    1/sum folded post-GEMM.
// ---------------------------------------------------------------------------
__global__ void av_kernel(const float* __restrict__ attn,
                          const float* __restrict__ rmax,
                          const float* __restrict__ rrcp,
                          const _Float16* __restrict__ vT,
                          _Float16* __restrict__ aT) {
  const int tid  = threadIdx.x;
  const int lane = tid & 31;
  int wid = (blockIdx.x << 3) + (tid >> 5);     // 0..4095
  const int mtile = wid & 63;
  const int bh    = wid >> 6;

  const int hi = lane >> 4, n = lane & 15;
  const int koffA = hi << 3, kbB = hi << 3;
  const int t0 = mtile << 4;
  const int rowA = t0 + (lane & 15);
  const float mA = rmax[((size_t)bh << 10) + rowA];
  const float* pa = attn + ((size_t)bh << 20) + (size_t)rowA * LL + koffA;
  const _Float16* vh = vT + ((size_t)bh << 6) * LL;
  const _Float16* bc0 = vh + (size_t)(     n) * LL + kbB;
  const _Float16* bc1 = vh + (size_t)(16 + n) * LL + kbB;
  const _Float16* bc2 = vh + (size_t)(32 + n) * LL + kbB;
  const _Float16* bc3 = vh + (size_t)(48 + n) * LL + kbB;

  union f8u { v8f v; float f[8]; };
  f8u r0, r1;                        // raw attn f32 operand, double-buffered
  r0.v = *(const v8f*)(pa);
  r1.v = *(const v8f*)(pa + 16);
  v8f acc0 = {}, acc1 = {}, acc2 = {}, acc3 = {};
  v16h B0 = load_b16(bc0), B1 = load_b16(bc1);
  v16h B2 = load_b16(bc2), B3 = load_b16(bc3);

  for (int kk = 0; kk < LL; kk += 32) {
    const int kn = (kk + 32 < LL) ? kk + 32 : kk;   // last iter: harmless reload
    f8u r0n, r1n;
    r0n.v = *(const v8f*)(pa + kn);
    r1n.v = *(const v8f*)(pa + kn + 16);
    v16h B0n = load_b16(bc0 + kn);
    v16h B1n = load_b16(bc1 + kn);
    v16h B2n = load_b16(bc2 + kn);
    v16h B3n = load_b16(bc3 + kn);

    union { v16h v; _Float16 e[16]; } uA;
#pragma unroll
    for (int i = 0; i < 8; ++i) {
      uA.e[i]     = (_Float16)__expf(r0.f[i] - mA);
      uA.e[8 + i] = (_Float16)__expf(r1.f[i] - mA);
    }
    acc0 = wmma16(uA.v, B0, acc0);
    acc1 = wmma16(uA.v, B1, acc1);
    acc2 = wmma16(uA.v, B2, acc2);
    acc3 = wmma16(uA.v, B3, acc3);
    r0 = r0n; r1 = r1n; B0 = B0n; B1 = B1n; B2 = B2n; B3 = B3n;
  }

  const int trow = t0 + (hi << 3);
  union f8u rc;
  rc.v = *(const v8f*)(rrcp + ((size_t)bh << 10) + trow);
  const int b = bh >> 3, hinb = bh & 7;
  _Float16* ab = aT + ((size_t)b << 10) * INNER + hinb * HD;

  f8u ua[4];
  ua[0].v = acc0; ua[1].v = acc1; ua[2].v = acc2; ua[3].v = acc3;
#pragma unroll
  for (int t = 0; t < 4; ++t)
#pragma unroll
    for (int r = 0; r < 8; ++r)
      ab[(size_t)(trow + r) * INNER + t * 16 + n] =
          (_Float16)(ua[t].f[r] * rc.f[r]);
}

// ---------------------------------------------------------------------------
// 6) out = Wo @ a + bo + x. Per wave: 16(M=o) x 64(N=l), K=512 (16 iters),
//    software pipelined.
// ---------------------------------------------------------------------------
__global__ void out_kernel(const float* __restrict__ Wo,
                           const float* __restrict__ bo,
                           const _Float16* __restrict__ aT,
                           const float* __restrict__ x,
                           float* __restrict__ out) {
  const int tid  = threadIdx.x;
  const int lane = tid & 31;
  int wid = (blockIdx.x << 3) + (tid >> 5);     // 0..4095
  const int ngrp  = wid & 15;
  const int mtile = (wid >> 4) & 31;
  const int b     = wid >> 9;

  const int m = lane & 15, hi = lane >> 4, n = lane & 15;
  const int koffA = hi << 3, kbB = hi << 3;
  const int l0 = ngrp << 6;
  const float* wrow = Wo + (size_t)(mtile * 16 + m) * INNER + koffA;
  const _Float16* ab = aT + ((size_t)b << 10) * INNER;
  const _Float16* bc0 = ab + (size_t)(l0      + n) * INNER + kbB;
  const _Float16* bc1 = ab + (size_t)(l0 + 16 + n) * INNER + kbB;
  const _Float16* bc2 = ab + (size_t)(l0 + 32 + n) * INNER + kbB;
  const _Float16* bc3 = ab + (size_t)(l0 + 48 + n) * INNER + kbB;

  v8f acc0 = {}, acc1 = {}, acc2 = {}, acc3 = {};
  v16h A  = load_a_cvt(wrow);
  v16h B0 = load_b16(bc0), B1 = load_b16(bc1);
  v16h B2 = load_b16(bc2), B3 = load_b16(bc3);
  for (int kk = 0; kk < INNER; kk += 32) {
    const int kn = (kk + 32 < INNER) ? kk + 32 : kk;
    v16h An  = load_a_cvt(wrow + kn);
    v16h B0n = load_b16(bc0 + kn);
    v16h B1n = load_b16(bc1 + kn);
    v16h B2n = load_b16(bc2 + kn);
    v16h B3n = load_b16(bc3 + kn);
    acc0 = wmma16(A, B0, acc0);
    acc1 = wmma16(A, B1, acc1);
    acc2 = wmma16(A, B2, acc2);
    acc3 = wmma16(A, B3, acc3);
    A = An; B0 = B0n; B1 = B1n; B2 = B2n; B3 = B3n;
  }

  const int obase = mtile * 16 + (hi << 3);
  union { v8f v; float f[8]; } b8;
  b8.v = *(const v8f*)(bo + obase);
  union { v8f v; float f[8]; } ua[4];
  ua[0].v = acc0; ua[1].v = acc1; ua[2].v = acc2; ua[3].v = acc3;
#pragma unroll
  for (int t = 0; t < 4; ++t) {
    const int l = l0 + t * 16 + n;
#pragma unroll
    for (int r = 0; r < 8; ++r) {
      const size_t idx = (((size_t)b * CC) + obase + r) * LL + l;
      out[idx] = ua[t].f[r] + b8.f[r] + x[idx];
    }
  }
}

// ---------------------------------------------------------------------------
extern "C" void kernel_launch(void* const* d_in, const int* in_sizes, int n_in,
                              void* d_out, int out_size, void* d_ws, size_t ws_size,
                              hipStream_t stream) {
  const float* x   = (const float*)d_in[0];
  // d_in[1] ("c") unused by the module
  const float* gg  = (const float*)d_in[2];
  const float* gb  = (const float*)d_in[3];
  const float* Wq  = (const float*)d_in[4];
  const float* bq  = (const float*)d_in[5];
  const float* Wk  = (const float*)d_in[6];
  const float* bk  = (const float*)d_in[7];
  const float* Wv  = (const float*)d_in[8];
  const float* bv  = (const float*)d_in[9];
  const float* Wo  = (const float*)d_in[10];
  const float* bo  = (const float*)d_in[11];

  float* out  = (float*)d_out;
  float* attn = out + (size_t)BB * CC * LL;     // attn_map after `out`

  char* ws = (char*)d_ws;
  _Float16* xnT  = (_Float16*)(ws);                          //  8 MB
  _Float16* q    = (_Float16*)(ws + ((size_t)8  << 20));     //  8 MB
  _Float16* k    = (_Float16*)(ws + ((size_t)16 << 20));     //  8 MB
  _Float16* v    = (_Float16*)(ws + ((size_t)24 << 20));     //  8 MB
  _Float16* aT   = (_Float16*)(ws + ((size_t)32 << 20));     //  8 MB
  float*    rmax = (float*)   (ws + ((size_t)40 << 20));     // 256 KB
  float*    rrcp = (float*)   (ws + ((size_t)40 << 20) + 65536 * sizeof(float));

  gn_kernel     <<<BB * 32,  256, 0, stream>>>(x, gg, gb, xnT);
  qkv_kernel    <<<1536,     256, 0, stream>>>(Wq, bq, Wk, bk, Wv, bv, xnT, q, k, v);
  qk_kernel     <<<8192,     256, 0, stream>>>(q, k, attn);
  rowstat_kernel<<<65536,    256, 0, stream>>>(attn, rmax, rrcp);
  av_kernel     <<<512,      256, 0, stream>>>(attn, rmax, rrcp, v, aT);
  out_kernel    <<<512,      256, 0, stream>>>(Wo, bo, aT, x, out);
}
